// MultiHeadSelfAttention_29729763622968
// MI455X (gfx1250) — matrix operand order repaired and checked by reading, not compiled, not run on hardware
//
#include <hip/hip_runtime.h>
#include <hip/hip_bf16.h>

// ---------------------------------------------------------------------------
// MHA for MI455X (gfx1250, wave32): bf16 WMMA, fp32 accumulate, async LDS.
//   proj_gemm  : Q/K/V = x @ W^T + b            (f32 in -> bf16 [B,H,S,dk])
//   flash      : causal flash-attention         (bf16 -> bf16 ctx [B,S,D])
//   out_gemm   : out = ctx @ W_o^T + b_o        (bf16 -> f32 [B,S,D])
// ---------------------------------------------------------------------------

typedef __attribute__((ext_vector_type(16))) __bf16 v16bf;
typedef __attribute__((ext_vector_type(8)))  __bf16 v8bf;
typedef __attribute__((ext_vector_type(8)))  float  v8f;
typedef __attribute__((ext_vector_type(4)))  int    v4i;

#define D_MODEL 1024
#define SEQ     2048
#define BATCH   4
#define HEADS   16
#define DK      64
#define TOKENS  (BATCH * SEQ)   // 8192

#if defined(__gfx1250__) && __has_builtin(__builtin_amdgcn_global_load_async_to_lds_b128)
#define USE_ASYNC_LDS 1
#else
#define USE_ASYNC_LDS 0
#endif

// 16-byte global -> LDS copy; async (ASYNCcnt) when available.
__device__ __forceinline__ void copy16_g2l(void* lds_dst, const void* gsrc) {
#if USE_ASYNC_LDS
  __builtin_amdgcn_global_load_async_to_lds_b128(
      (__attribute__((address_space(1))) v4i*)(uintptr_t)gsrc,
      (__attribute__((address_space(3))) v4i*)(uint32_t)(uintptr_t)lds_dst,
      0, 0);
#else
  *(v8bf*)lds_dst = *(const v8bf*)gsrc;
#endif
}

__device__ __forceinline__ void async_wait0() {
#if USE_ASYNC_LDS
  asm volatile("s_wait_asynccnt 0x0" ::: "memory");
#endif
}

__device__ __forceinline__ v8f wmma_bf16(v16bf a, v16bf b, v8f c) {
  return __builtin_amdgcn_wmma_f32_16x16x32_bf16(false, a, false, b, (short)0, c,
                                                 false, false);
}

// A fragment 16x32 bf16 (ISA 7.12.2): lane = row (lane&15); g = lane>>4:
//   elems 0..7 -> K = kk + g*8 + 0..7 ; elems 8..15 -> K = kk + 16 + g*8 + 0..7
__device__ __forceinline__ v16bf load_a_frag(const __bf16* base, int stride,
                                             int row, int kk, int lane) {
  const int g = lane >> 4;
  const __bf16* p = base + (size_t)row * stride + kk + g * 8;
  v8bf lo = *(const v8bf*)(p);
  v8bf hi = *(const v8bf*)(p + 16);
  v16bf a;
#pragma unroll
  for (int i = 0; i < 8; ++i) { a[i] = lo[i]; a[i + 8] = hi[i]; }
  return a;
}

// B fragment 32x16 bf16, LDS layout [col][k] (k contiguous):
//   lane = col (lane&15); g = lane>>4: elems 0..7 -> K = kk + g*8 + 0..7 ; elems 8..15 -> K = kk + 16 + g*8 + 0..7
__device__ __forceinline__ v16bf load_b_frag(const __bf16* base, int stride,
                                             int col, int kk, int lane) {
  const __bf16* p = base + (size_t)col * stride + kk + ((lane >> 4) << 3);
  v8bf lo = *(const v8bf*)(p);
  v8bf hi = *(const v8bf*)(p + 16);
  v16bf b;
#pragma unroll
  for (int i = 0; i < 8; ++i) { b[i] = lo[i]; b[i + 8] = hi[i]; }
  return b;
}

__device__ __forceinline__ v8f v8f_zero() {
  v8f z = {0.f, 0.f, 0.f, 0.f, 0.f, 0.f, 0.f, 0.f};
  return z;
}

// ---------------------------------------------------------------------------
// Projection GEMM: Out[b,h,s,d] = sum_k X[t,k]*W[n,k] + bias[n], bf16 out.
// 128x128 block tile, K-step 32, 8 waves (4x2), double-buffered LDS with
// register-staged prefetch (f32 source needs cvt, so no direct-to-LDS).
// ---------------------------------------------------------------------------
__global__ __launch_bounds__(256) void proj_gemm_kernel(
    const float* __restrict__ X, const float* __restrict__ W,
    const float* __restrict__ bias, __bf16* __restrict__ Out) {
  __shared__ alignas(16) __bf16 As[2][128][40];   // [buf][m][k]
  __shared__ alignas(16) __bf16 Bs[2][128][40];   // [buf][n][k]

  const int tid  = threadIdx.x;
  const int lane = tid & 31;
  const int wave = tid >> 5;
  const int wm   = wave & 3;
  const int wn   = wave >> 2;
  const int m0   = blockIdx.x * 128;
  const int n0   = blockIdx.y * 128;

  v8f acc[2][4];
#pragma unroll
  for (int mt = 0; mt < 2; ++mt)
#pragma unroll
    for (int nt = 0; nt < 4; ++nt) acc[mt][nt] = v8f_zero();

  float4 xr[4], wr[4];
  auto load_tiles = [&](int k0) {
#pragma unroll
    for (int it = 0; it < 4; ++it) {
      const int idx = tid + it * 256;
      const int row = idx >> 3, c4 = (idx & 7) * 4;
      xr[it] = *(const float4*)(X + (size_t)(m0 + row) * D_MODEL + k0 + c4);
      wr[it] = *(const float4*)(W + (size_t)(n0 + row) * D_MODEL + k0 + c4);
    }
  };
  auto store_tiles = [&](int b) {
#pragma unroll
    for (int it = 0; it < 4; ++it) {
      const int idx = tid + it * 256;
      const int row = idx >> 3, c4 = (idx & 7) * 4;
      As[b][row][c4 + 0] = (__bf16)xr[it].x; As[b][row][c4 + 1] = (__bf16)xr[it].y;
      As[b][row][c4 + 2] = (__bf16)xr[it].z; As[b][row][c4 + 3] = (__bf16)xr[it].w;
      Bs[b][row][c4 + 0] = (__bf16)wr[it].x; Bs[b][row][c4 + 1] = (__bf16)wr[it].y;
      Bs[b][row][c4 + 2] = (__bf16)wr[it].z; Bs[b][row][c4 + 3] = (__bf16)wr[it].w;
    }
  };

  load_tiles(0);
  store_tiles(0);
  __syncthreads();

  for (int k0 = 0; k0 < D_MODEL; k0 += 32) {
    const int  buf     = (k0 >> 5) & 1;
    const bool hasNext = (k0 + 32) < D_MODEL;
    if (hasNext) load_tiles(k0 + 32);          // global loads overlap WMMAs

    v16bf af[2];
#pragma unroll
    for (int mt = 0; mt < 2; ++mt)
      af[mt] = load_a_frag(&As[buf][0][0], 40, wm * 32 + mt * 16 + (lane & 15), 0, lane);
#pragma unroll
    for (int nt = 0; nt < 4; ++nt) {
      const v16bf bf_ = load_b_frag(&Bs[buf][0][0], 40, wn * 64 + nt * 16 + (lane & 15), 0, lane);
#pragma unroll
      for (int mt = 0; mt < 2; ++mt) acc[mt][nt] = wmma_bf16(af[mt], bf_, acc[mt][nt]);
    }
    if (hasNext) store_tiles(buf ^ 1);         // other buffer: no WAR hazard
    __syncthreads();
  }

#pragma unroll
  for (int nt = 0; nt < 4; ++nt) {
    const int n  = n0 + wn * 64 + nt * 16 + (lane & 15);
    const float bv = bias[n];
    const int h = n >> 6, d = n & 63;
#pragma unroll
    for (int mt = 0; mt < 2; ++mt) {
#pragma unroll
      for (int r = 0; r < 8; ++r) {
        const int m = m0 + wm * 32 + mt * 16 + ((lane >> 4) << 3) + r;  // token
        const int bb = m >> 11, s = m & (SEQ - 1);
        Out[(((size_t)bb * HEADS + h) * SEQ + s) * DK + d] = (__bf16)(acc[mt][nt][r] + bv);
      }
    }
  }
}

// ---------------------------------------------------------------------------
// Causal flash attention. Grid (S/128, B*H); 8 waves x 16 query rows.
// K/V tiles of 64 keys streamed via async global->LDS; V transposed LDS->LDS.
// ---------------------------------------------------------------------------
__global__ __launch_bounds__(256) void flash_kernel(
    const __bf16* __restrict__ Q, const __bf16* __restrict__ K,
    const __bf16* __restrict__ V, __bf16* __restrict__ ctx) {
  __shared__ alignas(16) __bf16 Qs[128][72];     // [q][d]
  __shared__ alignas(16) __bf16 Ks[64][72];      // [key][d]  (B-op for Q.K^T)
  __shared__ alignas(16) __bf16 Vn[64][72];      // [key][d]  natural V tile
  __shared__ alignas(16) __bf16 Vs[64][72];      // [d][key]  (B-op for P.V)
  __shared__ alignas(16) __bf16 Ps[8][16][72];   // per-wave P staging [row][key]

  const int tid  = threadIdx.x;
  const int lane = tid & 31;
  const int wave = tid >> 5;
  const int q0   = blockIdx.x * 128;
  const int bh   = blockIdx.y;                   // b*HEADS + h
  const int b    = bh >> 4, h = bh & 15;
  const size_t headOff = (size_t)bh * SEQ * DK;

  // Q tile 128x64 bf16, direct-to-LDS
#pragma unroll
  for (int it = 0; it < 4; ++it) {
    const int idx = tid + it * 256;
    const int row = idx >> 3, c8 = (idx & 7) * 8;
    copy16_g2l(&Qs[row][c8], Q + headOff + (size_t)(q0 + row) * DK + c8);
  }
  async_wait0();
  __syncthreads();

  const v16bf qa0 = load_a_frag(&Qs[0][0], 72, wave * 16 + (lane & 15), 0, lane);
  const v16bf qa1 = load_a_frag(&Qs[0][0], 72, wave * 16 + (lane & 15), 32, lane);

  float m_i[8], l_i[8];
  v8f o[4];
#pragma unroll
  for (int r = 0; r < 8; ++r) { m_i[r] = -1e30f; l_i[r] = 0.f; }
#pragma unroll
  for (int nt = 0; nt < 4; ++nt) o[nt] = v8f_zero();

  const int nTiles = 2 * blockIdx.x + 2;         // causal: keys up to q0+127
  for (int kt = 0; kt < nTiles; ++kt) {
    // issue K and V tile loads (previous readers finished at loop-end barrier)
#pragma unroll
    for (int it = 0; it < 2; ++it) {
      const int idx = tid + it * 256;
      const int row = idx >> 3, c8 = (idx & 7) * 8;
      const size_t g = headOff + (size_t)(kt * 64 + row) * DK + c8;
      copy16_g2l(&Ks[row][c8], K + g);
      copy16_g2l(&Vn[row][c8], V + g);
    }
    async_wait0();
    __syncthreads();

    // S = Q.K^T (16 x 64 per wave)
    v8f sc[4];
#pragma unroll
    for (int nt = 0; nt < 4; ++nt) {
      v8f s = v8f_zero();
      s = wmma_bf16(qa0, load_b_frag(&Ks[0][0], 72, nt * 16 + (lane & 15), 0, lane), s);
      s = wmma_bf16(qa1, load_b_frag(&Ks[0][0], 72, nt * 16 + (lane & 15), 32, lane), s);
      sc[nt] = s;
    }

    // scale + causal mask
    const int rowg0 = q0 + wave * 16 + ((lane >> 4) << 3);
#pragma unroll
    for (int nt = 0; nt < 4; ++nt) {
      const int col = kt * 64 + nt * 16 + (lane & 15);
#pragma unroll
      for (int r = 0; r < 8; ++r) {
        float sv = sc[nt][r] * 0.125f;                 // 1/sqrt(64)
        if (col > rowg0 + r) sv = -1e30f;
        sc[nt][r] = sv;
      }
    }

    // online softmax (row stats live in the 16-lane column group)
#pragma unroll
    for (int r = 0; r < 8; ++r) {
      float mx = fmaxf(fmaxf(sc[0][r], sc[1][r]), fmaxf(sc[2][r], sc[3][r]));
      mx = fmaxf(mx, __shfl_xor(mx, 1, 32));
      mx = fmaxf(mx, __shfl_xor(mx, 2, 32));
      mx = fmaxf(mx, __shfl_xor(mx, 4, 32));
      mx = fmaxf(mx, __shfl_xor(mx, 8, 32));
      const float mNew  = fmaxf(m_i[r], mx);
      const float alpha = __expf(m_i[r] - mNew);
      float ps = 0.f;
#pragma unroll
      for (int nt = 0; nt < 4; ++nt) {
        const float p = __expf(sc[nt][r] - mNew);
        sc[nt][r] = p;
        ps += p;
      }
      ps += __shfl_xor(ps, 1, 32);
      ps += __shfl_xor(ps, 2, 32);
      ps += __shfl_xor(ps, 4, 32);
      ps += __shfl_xor(ps, 8, 32);
      l_i[r] = l_i[r] * alpha + ps;
      m_i[r] = mNew;
#pragma unroll
      for (int nt = 0; nt < 4; ++nt) o[nt][r] *= alpha;
    }

    // stage P (C-layout -> A-layout) and transpose V (LDS->LDS, cheap ds ops)
#pragma unroll
    for (int nt = 0; nt < 4; ++nt)
#pragma unroll
      for (int r = 0; r < 8; ++r)
        Ps[wave][((lane >> 4) << 3) + r][nt * 16 + (lane & 15)] = (__bf16)sc[nt][r];
#pragma unroll
    for (int it = 0; it < 2; ++it) {
      const int idx = tid + it * 256;                  // 0..511
      const int key = idx >> 3, c8 = (idx & 7) * 8;
      const v8bf v = *(const v8bf*)&Vn[key][c8];
#pragma unroll
      for (int j = 0; j < 8; ++j) Vs[c8 + j][key] = v[j];
    }
    __syncthreads();

    // O += P.V
    const v16bf pa0 = load_a_frag(&Ps[wave][0][0], 72, lane & 15, 0, lane);
    const v16bf pa1 = load_a_frag(&Ps[wave][0][0], 72, lane & 15, 32, lane);
#pragma unroll
    for (int nt = 0; nt < 4; ++nt) {
      o[nt] = wmma_bf16(pa0, load_b_frag(&Vs[0][0], 72, nt * 16 + (lane & 15), 0, lane), o[nt]);
      o[nt] = wmma_bf16(pa1, load_b_frag(&Vs[0][0], 72, nt * 16 + (lane & 15), 32, lane), o[nt]);
    }
    __syncthreads();   // all Ks/Vn/Vs/Ps readers done before next iteration's writes
  }

  // normalize, store ctx bf16 [B,S,D] (token-major for final GEMM)
#pragma unroll
  for (int r = 0; r < 8; ++r) {
    const float inv = l_i[r] > 0.f ? 1.0f / l_i[r] : 0.f;
    const int srow = q0 + wave * 16 + ((lane >> 4) << 3) + r;
    const size_t base = ((size_t)b * SEQ + srow) * D_MODEL + h * DK;
#pragma unroll
    for (int nt = 0; nt < 4; ++nt)
      ctx[base + nt * 16 + (lane & 15)] = (__bf16)(o[nt][r] * inv);
  }
}

// ---------------------------------------------------------------------------
// Output GEMM: out[t,n] = sum_k ctx[t,k]*W_o[n,k] + b_o[n], fp32 out.
// A (bf16) tiles async direct-to-LDS, B (f32 W) register-staged; ping-pong.
// ---------------------------------------------------------------------------
__global__ __launch_bounds__(256) void out_gemm_kernel(
    const __bf16* __restrict__ Xb, const float* __restrict__ W,
    const float* __restrict__ bias, float* __restrict__ Out) {
  __shared__ alignas(16) __bf16 As[2][128][40];
  __shared__ alignas(16) __bf16 Bs[2][128][40];

  const int tid  = threadIdx.x;
  const int lane = tid & 31;
  const int wave = tid >> 5;
  const int wm   = wave & 3;
  const int wn   = wave >> 2;
  const int m0   = blockIdx.x * 128;
  const int n0   = blockIdx.y * 128;

  v8f acc[2][4];
#pragma unroll
  for (int mt = 0; mt < 2; ++mt)
#pragma unroll
    for (int nt = 0; nt < 4; ++nt) acc[mt][nt] = v8f_zero();

  float4 wr[4];
  auto issue_a = [&](int k0, int b) {
#pragma unroll
    for (int it = 0; it < 2; ++it) {
      const int idx = tid + it * 256;           // 0..511
      const int row = idx >> 2, c8 = (idx & 3) * 8;
      copy16_g2l(&As[b][row][c8], Xb + (size_t)(m0 + row) * D_MODEL + k0 + c8);
    }
  };
  auto load_b_regs = [&](int k0) {
#pragma unroll
    for (int it = 0; it < 4; ++it) {
      const int idx = tid + it * 256;
      const int row = idx >> 3, c4 = (idx & 7) * 4;
      wr[it] = *(const float4*)(W + (size_t)(n0 + row) * D_MODEL + k0 + c4);
    }
  };
  auto store_b = [&](int b) {
#pragma unroll
    for (int it = 0; it < 4; ++it) {
      const int idx = tid + it * 256;
      const int row = idx >> 3, c4 = (idx & 7) * 4;
      Bs[b][row][c4 + 0] = (__bf16)wr[it].x; Bs[b][row][c4 + 1] = (__bf16)wr[it].y;
      Bs[b][row][c4 + 2] = (__bf16)wr[it].z; Bs[b][row][c4 + 3] = (__bf16)wr[it].w;
    }
  };

  issue_a(0, 0);
  load_b_regs(0);
  store_b(0);
  async_wait0();
  __syncthreads();

  for (int k0 = 0; k0 < D_MODEL; k0 += 32) {
    const int  buf     = (k0 >> 5) & 1;
    const bool hasNext = (k0 + 32) < D_MODEL;
    if (hasNext) {
      issue_a(k0 + 32, buf ^ 1);               // async into other buffer
      load_b_regs(k0 + 32);
    }

    v16bf af[2];
#pragma unroll
    for (int mt = 0; mt < 2; ++mt)
      af[mt] = load_a_frag(&As[buf][0][0], 40, wm * 32 + mt * 16 + (lane & 15), 0, lane);
#pragma unroll
    for (int nt = 0; nt < 4; ++nt) {
      const v16bf bf_ = load_b_frag(&Bs[buf][0][0], 40, wn * 64 + nt * 16 + (lane & 15), 0, lane);
#pragma unroll
      for (int mt = 0; mt < 2; ++mt) acc[mt][nt] = wmma_bf16(af[mt], bf_, acc[mt][nt]);
    }
    if (hasNext) store_b(buf ^ 1);
    async_wait0();
    __syncthreads();
  }

#pragma unroll
  for (int nt = 0; nt < 4; ++nt) {
    const int n = n0 + wn * 64 + nt * 16 + (lane & 15);
    const float bv = bias[n];
#pragma unroll
    for (int mt = 0; mt < 2; ++mt) {
#pragma unroll
      for (int r = 0; r < 8; ++r) {
        const int m = m0 + wm * 32 + mt * 16 + ((lane >> 4) << 3) + r;
        Out[(size_t)m * D_MODEL + n] = acc[mt][nt][r] + bv;
      }
    }
  }
}

// ---------------------------------------------------------------------------
extern "C" void kernel_launch(void* const* d_in, const int* in_sizes, int n_in,
                              void* d_out, int out_size, void* d_ws, size_t ws_size,
                              hipStream_t stream) {
  const float* x   = (const float*)d_in[0];
  const float* W_q = (const float*)d_in[1];
  const float* b_q = (const float*)d_in[2];
  const float* W_k = (const float*)d_in[3];
  const float* b_k = (const float*)d_in[4];
  const float* W_v = (const float*)d_in[5];
  const float* b_v = (const float*)d_in[6];
  const float* W_o = (const float*)d_in[7];
  const float* b_o = (const float*)d_in[8];

  const size_t qkvElems = (size_t)TOKENS * D_MODEL;   // 8192*1024 bf16 each
  __bf16* Qb  = (__bf16*)d_ws;
  __bf16* Kb  = Qb + qkvElems;
  __bf16* Vb  = Kb + qkvElems;
  __bf16* ctx = Vb + qkvElems;                        // total 64 MB bf16

  dim3 gGemm(TOKENS / 128, D_MODEL / 128);            // (64, 8)
  proj_gemm_kernel<<<gGemm, 256, 0, stream>>>(x, W_q, b_q, Qb);
  proj_gemm_kernel<<<gGemm, 256, 0, stream>>>(x, W_k, b_k, Kb);
  proj_gemm_kernel<<<gGemm, 256, 0, stream>>>(x, W_v, b_v, Vb);

  flash_kernel<<<dim3(SEQ / 128, BATCH * HEADS), 256, 0, stream>>>(Qb, Kb, Vb, ctx);

  out_gemm_kernel<<<gGemm, 256, 0, stream>>>(ctx, W_o, b_o, (float*)d_out);
}
